// Attention_85237920956952
// MI455X (gfx1250) — matrix operand order repaired and checked by reading, not compiled, not run on hardware
//
#include <hip/hip_runtime.h>
#include <hip/hip_bf16.h>

typedef __bf16 bf16;
typedef __attribute__((ext_vector_type(16))) __bf16 v16bf;
typedef __attribute__((ext_vector_type(8)))  float  v8f;
typedef __attribute__((ext_vector_type(4)))  unsigned int v4u;
typedef __attribute__((ext_vector_type(8)))  int v8s;
typedef __attribute__((ext_vector_type(4)))  int v4s;

#define MOD 4
#define BB  2048
#define LL  1024
#define BL  ((size_t)BB * LL)          // 2,097,152
#define MBL ((size_t)MOD * BB * LL)    // 8,388,608
#define MLL ((size_t)MOD * LL * LL)    // 4,194,304
#define SCALE 0.03125f                 // 1/sqrt(1024)

// ---------------- CDNA5 data movement helpers ----------------------------------

// per-lane async global->LDS 16B copy (ASYNCcnt)
__device__ __forceinline__ void async_b128(void* lds, const void* g) {
  unsigned int l = (unsigned int)(size_t)lds;  // flat LDS aperture: addr[31:0] = LDS offset
  asm volatile("global_load_async_to_lds_b128 %0, %1, off" ::"v"(l), "v"(g)
               : "memory");
}
__device__ __forceinline__ void wait_async0() {
  asm volatile("s_wait_asynccnt 0x0" ::: "memory");
}

// Tensor Data Mover: 2D tile (tile_w x tile_h elements of 2 bytes) from global
// (row stride = stride_elems) into LDS. Issue from one wave; TENSORcnt tracked.
__device__ __forceinline__ void tdm_load_2d(void* lds, const void* gaddr,
                                            unsigned tile_w, unsigned tile_h,
                                            unsigned long long stride_elems) {
  unsigned long long ga = (unsigned long long)(size_t)gaddr;
  v4u g0;
  g0.x = 1u;                                   // count=1, is_restore=0
  g0.y = (unsigned)(size_t)lds;                // lds_addr
  g0.z = (unsigned)ga;                         // global_addr[31:0]
  g0.w = (unsigned)((ga >> 32) & 0x01ffffffull) | (2u << 30);  // addr hi | type=2
  v8s g1;
  g1[0] = (int)(1u << 16);                     // wg_mask=0, data_size=1 (2 bytes)
  g1[1] = (int)((tile_w & 0xffffu) << 16);     // tensor_dim0 = tile_w
  g1[2] = (int)((tile_w >> 16) | ((tile_h & 0xffffu) << 16));  // dim0 hi | tensor_dim1
  g1[3] = (int)((tile_h >> 16) | ((tile_w & 0xffffu) << 16));  // dim1 hi | tile_dim0
  g1[4] = (int)(tile_h & 0xffffu);             // tile_dim1 ; tile_dim2=0
  g1[5] = (int)(stride_elems & 0xffffffffu);   // tensor_dim0_stride lo
  g1[6] = (int)((stride_elems >> 32) & 0xffffu);
  g1[7] = 0;
  v4s z;
  z[0] = 0; z[1] = 0; z[2] = 0; z[3] = 0;
#if __clang_major__ >= 23
  v8s z8;
#pragma unroll
  for (int i = 0; i < 8; ++i) z8[i] = 0;
  __builtin_amdgcn_tensor_load_to_lds(g0, g1, z, z, z8, 0);
#else
  __builtin_amdgcn_tensor_load_to_lds(g0, g1, z, z, 0);
#endif
}
__device__ __forceinline__ void wait_tensor0() {
  __builtin_amdgcn_s_wait_tensorcnt(0);
}

// ---------------- WMMA fragment helpers (CDNA5 bf16 16x16x32 layouts) ----------

__device__ __forceinline__ v8f vzero8() {
  v8f z;
#pragma unroll
  for (int i = 0; i < 8; ++i) z[i] = 0.0f;
  return z;
}

// A-matrix 16x32 bf16: row = lane&15; lane<16 -> K {0..7,16..23}, else {8..15,24..31}
__device__ __forceinline__ v16bf load_A_frag(const bf16* base, int ld) {
  int lane = threadIdx.x & 31;
  const bf16* p = base + (size_t)(lane & 15) * ld + ((lane < 16) ? 0 : 8);
  v16bf a;
#pragma unroll
  for (int i = 0; i < 8; ++i) a[i] = p[i];
#pragma unroll
  for (int i = 0; i < 8; ++i) a[8 + i] = p[16 + i];
  return a;
}

// B-matrix 32x16 bf16 from K-major (B^T row-major [N][K]) storage
__device__ __forceinline__ v16bf load_B_frag(const bf16* baseT, int ld) {
  int lane = threadIdx.x & 31;
  const bf16* p = baseT + (size_t)(lane & 15) * ld + ((lane < 16) ? 0 : 8);
  v16bf b;
#pragma unroll
  for (int i = 0; i < 8; ++i) b[i] = p[i];
#pragma unroll
  for (int i = 0; i < 8; ++i) b[8 + i] = p[16 + i];
  return b;
}

__device__ __forceinline__ v8f wmma_bf16(v16bf a, v16bf b, v8f c) {
  return __builtin_amdgcn_wmma_f32_16x16x32_bf16(false, a, false, b, (short)0, c,
                                                 false, false);
}

// stage one 128x32 bf16 tile via async LDS loads (4 instrs / thread issue 2 here)
__device__ __forceinline__ void stage_tile_128x32(const bf16* gbase, int ldg,
                                                  bf16* lbase) {
  int t = threadIdx.x;
#pragma unroll
  for (int rep = 0; rep < 2; ++rep) {
    int idx = (t + rep * 256) * 8;
    int r = idx >> 5, c = idx & 31;
    async_b128(lbase + idx, gbase + (size_t)r * ldg + c);
  }
}

// ---------------- small prep kernels -------------------------------------------

__global__ __launch_bounds__(256) void k_convert(const float* __restrict__ in,
                                                 bf16* __restrict__ out, int n) {
  int i = blockIdx.x * 256 + threadIdx.x;
  if (i < n) out[i] = (bf16)in[i];
}

// out[s][c][r] = in[s][r][c]
__global__ __launch_bounds__(256) void k_transpose_bf16(const float* __restrict__ in,
                                                        bf16* __restrict__ out,
                                                        int S, int R, int C) {
  size_t total = (size_t)S * R * C;
  for (size_t i = (size_t)blockIdx.x * 256 + threadIdx.x; i < total;
       i += (size_t)gridDim.x * 256) {
    size_t rc = (size_t)R * C;
    size_t s = i / rc;
    size_t rem = i - s * rc;
    size_t c = rem / R;
    size_t r = rem - c * R;
    out[i] = (bf16)in[s * rc + r * C + c];
  }
}

__global__ __launch_bounds__(256) void k_prep_x(const float* __restrict__ x,
                                                bf16* __restrict__ xb,
                                                float* __restrict__ rowsum) {
  int row = blockIdx.x;  // m*B + b
  const float* src = x + (size_t)row * LL;
  bf16* dst = xb + (size_t)row * LL;
  float s = 0.0f;
  for (int i = threadIdx.x; i < LL; i += 256) {
    float v = src[i];
    s += v;
    dst[i] = (bf16)v;
  }
  __shared__ float red[256];
  red[threadIdx.x] = s;
  __syncthreads();
  for (int d = 128; d > 0; d >>= 1) {
    if (threadIdx.x < d) red[threadIdx.x] += red[threadIdx.x + d];
    __syncthreads();
  }
  if (threadIdx.x == 0) rowsum[row] = red[0];
}

__global__ __launch_bounds__(256) void k_scaler(const float* __restrict__ rowsum,
                                                float* __restrict__ scaler) {
  int b = blockIdx.x * 256 + threadIdx.x;
  if (b >= BB) return;
  int cnt = 0;
#pragma unroll
  for (int m = 0; m < MOD; ++m) cnt += (rowsum[m * BB + b] == 0.0f) ? 1 : 0;
  scaler[b] = (cnt > 0) ? (float)(cnt + 1) : 1.0f;
}

__global__ __launch_bounds__(256) void k_intra(const float* __restrict__ x,
                                               const float* __restrict__ aw,
                                               float* __restrict__ f_intra,
                                               bf16* __restrict__ gatein) {
  size_t idx = (size_t)blockIdx.x * 256 + threadIdx.x;
  if (idx >= BL) return;
  float a[MOD];
#pragma unroll
  for (int m = 0; m < MOD; ++m) a[m] = aw[m * BL + idx];
  float mx = a[0];
#pragma unroll
  for (int m = 1; m < MOD; ++m) mx = fmaxf(mx, a[m]);
  float e[MOD], s = 0.0f;
#pragma unroll
  for (int m = 0; m < MOD; ++m) { e[m] = __expf(a[m] - mx); s += e[m]; }
  float inv = 1.0f / s;
  float fi = 0.0f;
#pragma unroll
  for (int m = 0; m < MOD; ++m) fi += x[m * BL + idx] * e[m] * inv;
  f_intra[idx] = fi;
  size_t b = idx >> 10, l = idx & 1023;
  gatein[b * 2048 + l] = (bf16)fi;
}

__global__ __launch_bounds__(256) void k_cross_final(const float* __restrict__ fcp,
                                                     float* __restrict__ f_cross,
                                                     bf16* __restrict__ gatein) {
  size_t idx = (size_t)blockIdx.x * 256 + threadIdx.x;
  if (idx >= BL) return;
  float s = 0.0f;
#pragma unroll
  for (int m = 0; m < MOD; ++m) s += fcp[m * BL + idx];
  f_cross[idx] = s;
  size_t b = idx >> 10, l = idx & 1023;
  gatein[b * 2048 + 1024 + l] = (bf16)s;
}

// ---------------- generic bf16 WMMA GEMM: C = act(A @ Bt^T) --------------------
// async-LDS double-buffered staging; mode 0 tanh->f32, 1 ->bf16, 2 gate fusion
__global__ __launch_bounds__(256) void k_gemm(const bf16* __restrict__ A,
                                              const bf16* __restrict__ Bt, int K,
                                              int mode, float* __restrict__ outF,
                                              bf16* __restrict__ outB,
                                              const float* __restrict__ b_gate,
                                              const float* __restrict__ f_intra,
                                              const float* __restrict__ f_cross,
                                              const float* __restrict__ scaler,
                                              float* __restrict__ dout) {
  __shared__ bf16 sA[2][128 * 32];
  __shared__ bf16 sB[2][128 * 32];
  const int t = threadIdx.x;
  const int w = t >> 5;
  const int wr = w & 1;
  const int wc = w >> 1;
  const int rowBase = blockIdx.x * 128;
  const int colBase = blockIdx.y * 128;
  const bf16* Ab = A + (size_t)rowBase * K;
  const bf16* Bb = Bt + (size_t)colBase * K;

  v8f acc[4][2];
#pragma unroll
  for (int i = 0; i < 4; ++i) { acc[i][0] = vzero8(); acc[i][1] = vzero8(); }

  stage_tile_128x32(Ab, K, sA[0]);
  stage_tile_128x32(Bb, K, sB[0]);
  const int nk = K >> 5;
  for (int kk = 0; kk < nk; ++kk) {
    wait_async0();
    __syncthreads();
    int cur = kk & 1;
    if (kk + 1 < nk) {
      stage_tile_128x32(Ab + (kk + 1) * 32, K, sA[cur ^ 1]);
      stage_tile_128x32(Bb + (kk + 1) * 32, K, sB[cur ^ 1]);
    }
    v16bf bf0 = load_B_frag(&sB[cur][(wc * 32 + 0) * 32], 32);
    v16bf bf1 = load_B_frag(&sB[cur][(wc * 32 + 16) * 32], 32);
#pragma unroll
    for (int rt = 0; rt < 4; ++rt) {
      v16bf af = load_A_frag(&sA[cur][(wr * 64 + rt * 16) * 32], 32);
      acc[rt][0] = wmma_bf16(af, bf0, acc[rt][0]);
      acc[rt][1] = wmma_bf16(af, bf1, acc[rt][1]);
    }
  }

  const int lane = t & 31;
  const int colIn = lane & 15;
  const int rowOff = (lane < 16) ? 0 : 8;
#pragma unroll
  for (int rt = 0; rt < 4; ++rt) {
#pragma unroll
    for (int ct = 0; ct < 2; ++ct) {
      int rbase = rowBase + wr * 64 + rt * 16 + rowOff;
      int c = colBase + wc * 32 + ct * 16 + colIn;
#pragma unroll
      for (int i = 0; i < 8; ++i) {
        float v = acc[rt][ct][i];
        size_t r = (size_t)(rbase + i);
        if (mode == 0) {
          outF[r * LL + c] = tanhf(v);
        } else if (mode == 1) {
          outB[r * LL + c] = (bf16)v;
        } else {
          float g = 1.0f / (1.0f + __expf(-(v + b_gate[c])));
          float fi = f_intra[r * LL + c];
          float fc = f_cross[r * LL + c];
          dout[r * LL + c] = (g * fi + (1.0f - g) * fc) * scaler[r];
        }
      }
    }
  }
}

// ---------------- attention pass 1: per-(pair,row) max & sumexp of scaled S ----
__global__ __launch_bounds__(256) void k_attn_stats(const bf16* __restrict__ Qb,
                                                    const bf16* __restrict__ Kb,
                                                    float* __restrict__ smax,
                                                    float* __restrict__ ssum) {
  __shared__ bf16 sQ[16 * 1024];
  __shared__ bf16 sK[2][128 * 32];
  __shared__ float redM[8][16];
  __shared__ float redS[8][16];

  const int qt = blockIdx.x;
  const int p = blockIdx.y;
  const int m = p / 3;
  const int po = p % 3;
  const int o = po + ((po >= m) ? 1 : 0);
  const bf16* Q = Qb + (size_t)m * BL;
  const bf16* Kx = Kb + (size_t)o * BL;
  const int t = threadIdx.x;
  const int w = t >> 5;
  const int lane = t & 31;
  const int rowBase = qt * 16;

  // TDM: Q tile (16 contiguous rows) + first K tile; wave0 drives the DMA
  if (t < 32) {
    tdm_load_2d(sQ, Q + (size_t)rowBase * 1024, 1024, 16, 1024);
    tdm_load_2d(sK[0], Kx, 32, 128, 1024);
  }

  float rm[8], rs[8];
#pragma unroll
  for (int i = 0; i < 8; ++i) { rm[i] = -1e30f; rs[i] = 0.0f; }
  v8f acc = vzero8();

  for (int it = 0; it < 16 * 32; ++it) {
    if (t < 32) wait_tensor0();
    __syncthreads();
    int cur = it & 1;
    if (it + 1 < 16 * 32 && t < 32) {
      int nit = it + 1;
      size_t nc0 = (size_t)(nit >> 5) * 128;
      int nks = (nit & 31) * 32;
      tdm_load_2d(sK[cur ^ 1], Kx + nc0 * 1024 + nks, 32, 128, 1024);
    }
    int ks = it & 31;
    v16bf af = load_A_frag(&sQ[ks * 32], 1024);
    v16bf bf = load_B_frag(&sK[cur][(w * 16) * 32], 32);
    acc = wmma_bf16(af, bf, acc);

    if (ks == 31) {  // chunk done: fold 16-col tile into running row stats
      float mx[8];
#pragma unroll
      for (int i = 0; i < 8; ++i) mx[i] = acc[i] * SCALE;
#pragma unroll
      for (int d = 1; d < 16; d <<= 1)
#pragma unroll
        for (int i = 0; i < 8; ++i) mx[i] = fmaxf(mx[i], __shfl_xor(mx[i], d, 32));
      float se[8];
#pragma unroll
      for (int i = 0; i < 8; ++i) se[i] = __expf(acc[i] * SCALE - mx[i]);
#pragma unroll
      for (int d = 1; d < 16; d <<= 1)
#pragma unroll
        for (int i = 0; i < 8; ++i) se[i] += __shfl_xor(se[i], d, 32);
#pragma unroll
      for (int i = 0; i < 8; ++i) {
        float nm = fmaxf(rm[i], mx[i]);
        rs[i] = rs[i] * __expf(rm[i] - nm) + se[i] * __expf(mx[i] - nm);
        rm[i] = nm;
      }
      acc = vzero8();
    }
  }

  if (lane == 0 || lane == 16) {
    int half = lane >> 4;
#pragma unroll
    for (int i = 0; i < 8; ++i) {
      redM[w][half * 8 + i] = rm[i];
      redS[w][half * 8 + i] = rs[i];
    }
  }
  __syncthreads();
  if (t < 16) {
    float M = -1e30f;
#pragma unroll
    for (int w2 = 0; w2 < 8; ++w2) M = fmaxf(M, redM[w2][t]);
    float S = 0.0f;
#pragma unroll
    for (int w2 = 0; w2 < 8; ++w2) S += redS[w2][t] * __expf(redM[w2][t] - M);
    smax[(size_t)p * BB + rowBase + t] = M;
    ssum[(size_t)p * BB + rowBase + t] = S;
  }
}

// ---------------- attention pass 2: O += softmax(S) @ X_o, summed over o!=m ----
#define ACC_SMEM (32768 + 4096 + 2 * 8192 + 2 * 65536)
__global__ __launch_bounds__(256) void k_attn_accum(const bf16* __restrict__ Qb,
                                                    const bf16* __restrict__ Kb,
                                                    const bf16* __restrict__ xT,
                                                    const float* __restrict__ smax,
                                                    const float* __restrict__ ssum,
                                                    float* __restrict__ fcp) {
  extern __shared__ char smem[];
  bf16* sQ = (bf16*)smem;                            // 16*1024
  bf16* sP = (bf16*)(smem + 32768);                  // 16*128
  bf16* sK0 = (bf16*)(smem + 32768 + 4096);          // 128*32 x2
  bf16* sK1 = sK0 + 128 * 32;
  bf16* sX0 = (bf16*)(smem + 32768 + 4096 + 16384);  // 1024*32 x2
  bf16* sX1 = sX0 + 1024 * 32;

  const int qt = blockIdx.x;
  const int m = blockIdx.y;
  const int t = threadIdx.x;
  const int w = t >> 5;
  const int lane = t & 31;
  const int colIn = lane & 15;
  const int rowOff = (lane < 16) ? 0 : 8;
  const int rowBase = qt * 16;

  const bf16* Q = Qb + (size_t)m * BL;
  if (t < 32) tdm_load_2d(sQ, Q + (size_t)rowBase * 1024, 1024, 16, 1024);

  v8f O[8];
#pragma unroll
  for (int i = 0; i < 8; ++i) O[i] = vzero8();

  for (int oo = 0; oo < 3; ++oo) {
    int o = oo + ((oo >= m) ? 1 : 0);
    int p = m * 3 + (o - ((o > m) ? 1 : 0));
    const bf16* Kx = Kb + (size_t)o * BL;
    const bf16* XT = xT + (size_t)o * BL;  // [1024 l][2048 b]
    float rm[8], rinv[8];
#pragma unroll
    for (int i = 0; i < 8; ++i) {
      int r = rowBase + rowOff + i;
      rm[i] = smax[(size_t)p * BB + r];
      rinv[i] = 1.0f / ssum[(size_t)p * BB + r];
    }

    for (int chunk = 0; chunk < 16; ++chunk) {
      size_t c0 = (size_t)chunk * 128;
      // ---- phase A: S tile (16 x 128), each wave 16 cols, K double-buffered
      if (t < 32) tdm_load_2d(sK0, Kx + c0 * 1024, 32, 128, 1024);
      v8f acc = vzero8();
      for (int ks = 0; ks < 32; ++ks) {
        bf16* cur = (ks & 1) ? sK1 : sK0;
        bf16* nxt = (ks & 1) ? sK0 : sK1;
        if (t < 32) wait_tensor0();
        __syncthreads();
        if (ks < 31 && t < 32)
          tdm_load_2d(nxt, Kx + c0 * 1024 + (ks + 1) * 32, 32, 128, 1024);
        v16bf af = load_A_frag(&sQ[ks * 32], 1024);
        v16bf bf = load_B_frag(&cur[(w * 16) * 32], 32);
        acc = wmma_bf16(af, bf, acc);
      }
      // ---- P = exp(scaled S - max) / sum -> LDS
#pragma unroll
      for (int i = 0; i < 8; ++i) {
        float pv = __expf(acc[i] * SCALE - rm[i]) * rinv[i];
        sP[(rowOff + i) * 128 + w * 16 + colIn] = (bf16)pv;
      }
      __syncthreads();
      // ---- phase B: O += P @ X_o^T-slice, X^T double-buffered via TDM
      if (t < 32) tdm_load_2d(sX0, XT + c0, 32, 1024, 2048);
      for (int cs = 0; cs < 4; ++cs) {
        bf16* cur = (cs & 1) ? sX1 : sX0;
        bf16* nxt = (cs & 1) ? sX0 : sX1;
        if (t < 32) wait_tensor0();
        __syncthreads();
        if (cs < 3 && t < 32) tdm_load_2d(nxt, XT + c0 + (cs + 1) * 32, 32, 1024, 2048);
        v16bf pa = load_A_frag(&sP[cs * 32], 128);
#pragma unroll
        for (int nt = 0; nt < 8; ++nt) {
          v16bf xb = load_B_frag(&cur[(w * 128 + nt * 16) * 32], 32);
          O[nt] = wmma_bf16(pa, xb, O[nt]);
        }
      }
    }
  }

#pragma unroll
  for (int nt = 0; nt < 8; ++nt) {
#pragma unroll
    for (int i = 0; i < 8; ++i) {
      size_t r = (size_t)rowBase + rowOff + i;
      int c = w * 128 + nt * 16 + colIn;
      fcp[((size_t)m * BB + r) * LL + c] = O[nt][i] * 0.25f;  // mean over m
    }
  }
}

// ---------------- host-side orchestration --------------------------------------

extern "C" void kernel_launch(void* const* d_in, const int* in_sizes, int n_in,
                              void* d_out, int out_size, void* d_ws, size_t ws_size,
                              hipStream_t stream) {
  const float* x = (const float*)d_in[0];
  const float* Wp = (const float*)d_in[1];
  const float* Wa = (const float*)d_in[2];
  const float* Wg = (const float*)d_in[3];
  const float* bg = (const float*)d_in[4];
  float* out = (float*)d_out;

  char* ws = (char*)d_ws;
  size_t off = 0;
  auto alloc = [&](size_t bytes) -> char* {
    char* p = ws + off;
    off += (bytes + 255) & ~(size_t)255;
    return p;
  };
  bf16* x_bf = (bf16*)alloc(MBL * 2);
  bf16* xT_bf = (bf16*)alloc(MBL * 2);
  bf16* Q_bf = (bf16*)alloc(MBL * 2);
  bf16* K_bf = (bf16*)alloc(MBL * 2);
  bf16* Wp_bf = (bf16*)alloc(MLL * 2);
  bf16* Wa_bf = (bf16*)alloc(MLL * 2);
  bf16* WaT_bf = (bf16*)alloc(MLL * 2);
  bf16* Wg_bf = (bf16*)alloc((size_t)LL * 2 * LL * 2);
  float* aw = (float*)alloc(MBL * 4);  // reused as fcp after k_intra
  bf16* gatein = (bf16*)alloc((size_t)BB * 2 * LL * 2);
  float* f_intra = (float*)alloc(BL * 4);
  float* f_cross = (float*)alloc(BL * 4);
  float* rowsum = (float*)alloc(MOD * BB * 4);
  float* scaler = (float*)alloc(BB * 4);
  float* smax = (float*)alloc(12 * BB * 4);
  float* ssum = (float*)alloc(12 * BB * 4);
  if (off > ws_size) return;
  float* fcp = aw;

  k_convert<<<(int)((MLL + 255) / 256), 256, 0, stream>>>(Wp, Wp_bf, (int)MLL);
  k_convert<<<(int)((MLL + 255) / 256), 256, 0, stream>>>(Wa, Wa_bf, (int)MLL);
  k_convert<<<(int)((LL * 2 * LL + 255) / 256), 256, 0, stream>>>(Wg, Wg_bf,
                                                                  LL * 2 * LL);
  k_transpose_bf16<<<(int)(MLL / 256), 256, 0, stream>>>(Wa, WaT_bf, MOD, LL, LL);
  k_transpose_bf16<<<(int)(MBL / 256), 256, 0, stream>>>(x, xT_bf, MOD, BB, LL);
  k_prep_x<<<MOD * BB, 256, 0, stream>>>(x, x_bf, rowsum);
  k_scaler<<<(BB + 255) / 256, 256, 0, stream>>>(rowsum, scaler);

  dim3 ggrid(16, 8);
  for (int m = 0; m < MOD; ++m) {
    k_gemm<<<ggrid, 256, 0, stream>>>(x_bf + m * BL, Wp_bf + (size_t)m * LL * LL,
                                      LL, 0, aw + m * BL, nullptr, nullptr,
                                      nullptr, nullptr, nullptr, nullptr);
    k_gemm<<<ggrid, 256, 0, stream>>>(x_bf + m * BL, WaT_bf + (size_t)m * LL * LL,
                                      LL, 1, nullptr, Q_bf + m * BL, nullptr,
                                      nullptr, nullptr, nullptr, nullptr);
    k_gemm<<<ggrid, 256, 0, stream>>>(x_bf + m * BL, Wa_bf + (size_t)m * LL * LL,
                                      LL, 1, nullptr, K_bf + m * BL, nullptr,
                                      nullptr, nullptr, nullptr, nullptr);
  }

  k_intra<<<(int)(BL / 256), 256, 0, stream>>>(x, aw, f_intra, gatein);

  k_attn_stats<<<dim3(128, 12), 256, 0, stream>>>(Q_bf, K_bf, smax, ssum);
  k_attn_accum<<<dim3(128, 4), 256, ACC_SMEM, stream>>>(Q_bf, K_bf, xT_bf, smax,
                                                        ssum, fcp);
  k_cross_final<<<(int)(BL / 256), 256, 0, stream>>>(fcp, f_cross, gatein);

  k_gemm<<<ggrid, 256, 0, stream>>>(gatein, Wg_bf, 2 * LL, 2, nullptr, nullptr,
                                    bg, f_intra, f_cross, scaler, out);
}
